// S5VH_Model_77721728188536
// MI455X (gfx1250) — matrix operand order repaired and checked by reading, not compiled, not run on hardware
//
#include <hip/hip_runtime.h>
#include <hip/hip_bf16.h>
#include <math.h>

// ---------------------------------------------------------------------------
// S5VH forward for MI455X (gfx1250, wave32).  All dense projections use
// v_wmma_f32_16x16x32_f16 (f16 operands, f32 accumulate) with vectorized
// b128 fragment loads (K%32==0 enforced; dt_proj K=16 is zero-padded to 32).
// Each wave computes a 16x64 C tile; branch-free 4-WMMA inner loop for the
// full-tile case, generic loop only for the N=48 x_proj.
// ---------------------------------------------------------------------------

typedef __attribute__((ext_vector_type(16))) _Float16 v16h;
typedef __attribute__((ext_vector_type(8)))  float    v8f;

#define BATCH    64
#define FEAT     4096
#define D_MODEL  256
#define D_INNER  512
#define D_STATE  16
#define DT_RANK  16
#define D_CONV   4
#define MAXT     25
#define N_VIS    12
#define N_MASK   13
#define NBITS    64
#define ENC_D    12
#define DEC_D    8
#define LN_EPS   1e-5f

// --------------------------- wave reduction --------------------------------
__device__ __forceinline__ float wave_sum32(float v) {
  #pragma unroll
  for (int off = 16; off > 0; off >>= 1) v += __shfl_xor(v, off, 32);
  return v;
}

// --------------------------- fragment loaders ------------------------------
// A 16x32 fragment, this code's own contiguous k order (matches B): lanes 0-15 hold K 0..15 of their row;
// lanes 16-31 hold K 16..31.  Contiguous 16-float run.
__device__ __forceinline__ v16h load_a_frag(const float* __restrict__ Ar,
                                            int k0, int half) {
  const float* p0 = Ar + k0 + half * 16;       // 8 floats
  const float* p1 = p0 + 8;                    // 8 floats
  float4 c0 = *(const float4*)(p0);
  float4 c1 = *(const float4*)(p0 + 4);
  float4 c2 = *(const float4*)(p1);
  float4 c3 = *(const float4*)(p1 + 4);
  v16h a;
  a[0]  = (_Float16)c0.x; a[1]  = (_Float16)c0.y;
  a[2]  = (_Float16)c0.z; a[3]  = (_Float16)c0.w;
  a[4]  = (_Float16)c1.x; a[5]  = (_Float16)c1.y;
  a[6]  = (_Float16)c1.z; a[7]  = (_Float16)c1.w;
  a[8]  = (_Float16)c2.x; a[9]  = (_Float16)c2.y;
  a[10] = (_Float16)c2.z; a[11] = (_Float16)c2.w;
  a[12] = (_Float16)c3.x; a[13] = (_Float16)c3.y;
  a[14] = (_Float16)c3.z; a[15] = (_Float16)c3.w;
  return a;
}

// B 32x16 fragment, same contiguous k order as A: lanes 0-15 hold K 0..15 of their column, lanes
// 16-31 hold K 16..31.  B col c == W row c (K-contiguous), 16-float run.
__device__ __forceinline__ v16h load_b_frag(const float* __restrict__ Wr,
                                            int k0, int half) {
  const float* p = Wr + k0 + half * 16;        // 16 floats
  float4 c0 = *(const float4*)(p);
  float4 c1 = *(const float4*)(p + 4);
  float4 c2 = *(const float4*)(p + 8);
  float4 c3 = *(const float4*)(p + 12);
  v16h b;
  b[0]  = (_Float16)c0.x; b[1]  = (_Float16)c0.y;
  b[2]  = (_Float16)c0.z; b[3]  = (_Float16)c0.w;
  b[4]  = (_Float16)c1.x; b[5]  = (_Float16)c1.y;
  b[6]  = (_Float16)c1.z; b[7]  = (_Float16)c1.w;
  b[8]  = (_Float16)c2.x; b[9]  = (_Float16)c2.y;
  b[10] = (_Float16)c2.z; b[11] = (_Float16)c2.w;
  b[12] = (_Float16)c3.x; b[13] = (_Float16)c3.y;
  b[14] = (_Float16)c3.z; b[15] = (_Float16)c3.w;
  return b;
}

// --------------------------- epilogue (direct acc regs, no arrays) ---------
__device__ __forceinline__ void epilogue_store(float* __restrict__ C,
                                               v8f acc, int N, int rbase,
                                               int col,
                                               const float* __restrict__ bias,
                                               const float* __restrict__ aux,
                                               int auxT, int mode) {
  const float bv = bias ? bias[col] : 0.f;
  #pragma unroll
  for (int j = 0; j < 8; ++j) {
    const int row = rbase + j;
    float v = acc[j];
    if (mode == 0) {
      v += bv;
    } else if (mode == 1) {                       // softplus
      v += bv;
      v = (v > 20.f) ? v : log1pf(__expf(v));
    } else if (mode == 2) {                       // hash: rint(tanh(.))
      v += bv;
      v = rintf(tanhf(v));
    } else {                                      // patch embed: + pos
      v += bv + aux[(size_t)(row % auxT) * N + col];
    }
    C[(size_t)row * N + col] = v;
  }
}

// --------------------------- generic WMMA GEMM -----------------------------
// C[M,N] = A[M,K] (row-major, lda) * W[N,K]^T (row-major, ldw=K) + epilogue
// REQUIRES: K % 32 == 0, M % 16 == 0, N % 16 == 0, 16B-aligned rows.
// epilogue modes: 0: +bias   1: softplus(x+bias)   2: rint(tanh(x+bias))
//                 3: +bias + aux[(row % auxT)*N + col]   (patch embed + pos)
// One wave computes a 16x64 tile (4 accums); 4 waves/block; z batches dirs.
__global__ __launch_bounds__(128)
void gemm_wmma_kernel(const float* __restrict__ A, int lda,
                      const float* __restrict__ W,
                      float* __restrict__ C,
                      int M, int N, int K,
                      long sA, long sW, long sC, long sBias,
                      const float* __restrict__ bias,
                      const float* __restrict__ aux, int auxT,
                      int mode) {
  const int z = blockIdx.z;
  A += (size_t)z * sA;
  W += (size_t)z * sW;
  C += (size_t)z * sC;
  if (bias) bias += (size_t)z * sBias;

  const int wave = threadIdx.x >> 5;
  const int lane = threadIdx.x & 31;
  const int tileM = blockIdx.y * 4 + wave;
  if (tileM * 16 >= M) return;              // uniform per wave (no barriers)

  const int row0 = tileM * 16;
  const int col0 = blockIdx.x * 64;
  const int nT = ((N - col0) >= 64) ? 4 : ((N - col0) >> 4);  // uniform

  const int l16  = lane & 15;
  const int half = lane >> 4;

  const float* Ar  = A + (size_t)(row0 + l16) * lda;
  const float* Wr0 = W + (size_t)(col0 + l16) * K;       // B tile 0
  const float* Wr1 = Wr0 + (size_t)16 * K;
  const float* Wr2 = Wr0 + (size_t)32 * K;
  const float* Wr3 = Wr0 + (size_t)48 * K;

  v8f acc0 = {}, acc1 = {}, acc2 = {}, acc3 = {};

  if (nT == 4) {
    // branch-free hot loop: 1 A fragment + 4 B fragments -> 4 WMMAs
    for (int k0 = 0; k0 < K; k0 += 32) {
      const v16h afr = load_a_frag(Ar, k0, half);
      const v16h b0 = load_b_frag(Wr0, k0, half);
      const v16h b1 = load_b_frag(Wr1, k0, half);
      const v16h b2 = load_b_frag(Wr2, k0, half);
      const v16h b3 = load_b_frag(Wr3, k0, half);
      acc0 = __builtin_amdgcn_wmma_f32_16x16x32_f16(false, afr, false, b0,
                                                    (short)0, acc0, false, false);
      acc1 = __builtin_amdgcn_wmma_f32_16x16x32_f16(false, afr, false, b1,
                                                    (short)0, acc1, false, false);
      acc2 = __builtin_amdgcn_wmma_f32_16x16x32_f16(false, afr, false, b2,
                                                    (short)0, acc2, false, false);
      acc3 = __builtin_amdgcn_wmma_f32_16x16x32_f16(false, afr, false, b3,
                                                    (short)0, acc3, false, false);
    }
  } else {
    // partial tile (only x_proj N=48 reaches this): uniform per-wave count
    for (int k0 = 0; k0 < K; k0 += 32) {
      const v16h afr = load_a_frag(Ar, k0, half);
      {
        v16h b = load_b_frag(Wr0, k0, half);
        acc0 = __builtin_amdgcn_wmma_f32_16x16x32_f16(false, afr, false, b,
                                                      (short)0, acc0, false, false);
      }
      if (nT > 1) {
        v16h b = load_b_frag(Wr1, k0, half);
        acc1 = __builtin_amdgcn_wmma_f32_16x16x32_f16(false, afr, false, b,
                                                      (short)0, acc1, false, false);
      }
      if (nT > 2) {
        v16h b = load_b_frag(Wr2, k0, half);
        acc2 = __builtin_amdgcn_wmma_f32_16x16x32_f16(false, afr, false, b,
                                                      (short)0, acc2, false, false);
      }
    }
  }

  // C layout: lane holds column (tile base + l16), rows row0 + 8*half + j
  const int rbase = row0 + 8 * half;
  epilogue_store(C, acc0, N, rbase, col0 + l16, bias, aux, auxT, mode);
  if (nT > 1) epilogue_store(C, acc1, N, rbase, col0 + 16 + l16, bias, aux, auxT, mode);
  if (nT > 2) epilogue_store(C, acc2, N, rbase, col0 + 32 + l16, bias, aux, auxT, mode);
  if (nT > 3) epilogue_store(C, acc3, N, rbase, col0 + 48 + l16, bias, aux, auxT, mode);
}

// --------------------------- LayerNorm (dual, with flip) -------------------
__global__ void ln_dual_kernel(const float* __restrict__ x,
                               float* __restrict__ u,
                               const float* __restrict__ lnw,
                               const float* __restrict__ lnb,
                               int B, int L) {
  const int wid  = (blockIdx.x * blockDim.x + threadIdx.x) >> 5;
  const int lane = threadIdx.x & 31;
  const int total = 2 * B * L;
  if (wid >= total) return;
  const int dir = wid / (B * L);
  const int bt  = wid % (B * L);
  const int b = bt / L, t = bt % L;
  const int ts = dir ? (L - 1 - t) : t;
  const float* row = x + ((size_t)b * L + ts) * D_MODEL;

  float vals[8], s = 0.f;
  #pragma unroll
  for (int i = 0; i < 8; ++i) { vals[i] = row[lane + 32 * i]; s += vals[i]; }
  const float mean = wave_sum32(s) * (1.f / D_MODEL);
  float vs = 0.f;
  #pragma unroll
  for (int i = 0; i < 8; ++i) { float d = vals[i] - mean; vs += d * d; }
  const float rstd = rsqrtf(wave_sum32(vs) * (1.f / D_MODEL) + LN_EPS);

  const float* w  = lnw + dir * D_MODEL;
  const float* bb = lnb + dir * D_MODEL;
  float* out = u + (size_t)wid * D_MODEL;
  #pragma unroll
  for (int i = 0; i < 8; ++i) {
    int c = lane + 32 * i;
    out[c] = (vals[i] - mean) * rstd * w[c] + bb[c];
  }
}

// --------------------------- LayerNorm (row remap) -------------------------
__global__ void ln_map_kernel(const float* __restrict__ in,
                              float* __restrict__ out,
                              const float* __restrict__ w,
                              const float* __restrict__ bb,
                              int B, int inL, int outL, int tOff) {
  const int wid  = (blockIdx.x * blockDim.x + threadIdx.x) >> 5;
  const int lane = threadIdx.x & 31;
  if (wid >= B * outL) return;
  const int b = wid / outL, t = wid % outL;
  const float* row = in + ((size_t)b * inL + tOff + t) * D_MODEL;

  float vals[8], s = 0.f;
  #pragma unroll
  for (int i = 0; i < 8; ++i) { vals[i] = row[lane + 32 * i]; s += vals[i]; }
  const float mean = wave_sum32(s) * (1.f / D_MODEL);
  float vs = 0.f;
  #pragma unroll
  for (int i = 0; i < 8; ++i) { float d = vals[i] - mean; vs += d * d; }
  const float rstd = rsqrtf(wave_sum32(vs) * (1.f / D_MODEL) + LN_EPS);

  float* o = out + (size_t)wid * D_MODEL;
  #pragma unroll
  for (int i = 0; i < 8; ++i) {
    int c = lane + 32 * i;
    o[c] = (vals[i] - mean) * rstd * w[c] + bb[c];
  }
}

// --------------------------- causal conv (width 4) + SiLU ------------------
__global__ void conv_silu_kernel(const float* __restrict__ xz,
                                 float* __restrict__ xs,
                                 const float* __restrict__ cw,
                                 const float* __restrict__ cb,
                                 int B, int L) {
  const size_t total = (size_t)2 * B * L * D_INNER;
  const size_t i = (size_t)blockIdx.x * blockDim.x + threadIdx.x;
  if (i >= total) return;
  const int d = (int)(i % D_INNER);
  const size_t r = i / D_INNER;             // dir*B*L + b*L + t
  const int t = (int)(r % L);
  const int dir = (int)(r / ((size_t)B * L));
  const float* cwd = cw + (size_t)dir * D_INNER * D_CONV + (size_t)d * D_CONV;
  float acc = cb[dir * D_INNER + d];
  #pragma unroll
  for (int j = 0; j < D_CONV; ++j) {
    int tt = t + j - (D_CONV - 1);
    if (tt >= 0) acc += cwd[j] * xz[(r + (size_t)(tt - t)) * (2 * D_INNER) + d];
  }
  xs[i] = acc / (1.f + __expf(-acc));       // silu
}

// --------------------------- pad dt_raw (48-col -> 32-col zero-padded) -----
__global__ void pad_dt_kernel(const float* __restrict__ dbl,
                              float* __restrict__ dtpad, int rows) {
  const size_t total = (size_t)rows * 32;
  const size_t i = (size_t)blockIdx.x * blockDim.x + threadIdx.x;
  if (i >= total) return;
  const int c = (int)(i & 31);
  const size_t r = i >> 5;
  dtpad[i] = (c < DT_RANK) ? dbl[r * (DT_RANK + 2 * D_STATE) + c] : 0.f;
}

// --------------------------- selective scan --------------------------------
__global__ void scan_kernel(const float* __restrict__ dt,
                            const float* __restrict__ xs,
                            const float* __restrict__ dbl,
                            const float* __restrict__ xz,
                            const float* __restrict__ Al,
                            const float* __restrict__ Dd,
                            float* __restrict__ ycomb,
                            int B, int L) {
  const int i = blockIdx.x * blockDim.x + threadIdx.x;
  if (i >= 2 * B * D_INNER) return;
  const int d   = i % D_INNER;
  const int db  = i / D_INNER;
  const int dir = db / B;
  const int b   = db % B;

  float Aa[D_STATE];
  const float* Alp = Al + ((size_t)dir * D_INNER + d) * D_STATE;
  #pragma unroll
  for (int n = 0; n < D_STATE; ++n) Aa[n] = -__expf(Alp[n]);
  const float Dv = Dd[dir * D_INNER + d];

  float h[D_STATE];
  #pragma unroll
  for (int n = 0; n < D_STATE; ++n) h[n] = 0.f;

  const size_t rowbase = ((size_t)dir * B + b) * L;
  for (int t = 0; t < L; ++t) {
    const size_t r = rowbase + t;
    const float dtv = dt[r * D_INNER + d];
    const float xv  = xs[r * D_INNER + d];
    const float* dr = dbl + r * (DT_RANK + 2 * D_STATE);
    float y = 0.f;
    #pragma unroll
    for (int n = 0; n < D_STATE; ++n) {
      const float dA = __expf(dtv * Aa[n]);
      h[n] = dA * h[n] + dtv * dr[DT_RANK + n] * xv;
      y += h[n] * dr[DT_RANK + D_STATE + n];
    }
    const float zz = xz[r * (2 * D_INNER) + D_INNER + d];
    const float sz = zz / (1.f + __expf(-zz));
    ycomb[r * D_INNER + d] = (y + xv * Dv) * sz;
  }
}

// --------------------------- residual combine (with flip of dir1) ----------
__global__ void combine_kernel(float* __restrict__ x,
                               const float* __restrict__ outs,
                               int B, int L) {
  const size_t total = (size_t)B * L * D_MODEL;
  const size_t i = (size_t)blockIdx.x * blockDim.x + threadIdx.x;
  if (i >= total) return;
  const int d = (int)(i % D_MODEL);
  const size_t r = i / D_MODEL;
  const int t = (int)(r % L);
  const size_t b = r / L;
  const float v0 = outs[((size_t)0 * B * L + b * L + t) * D_MODEL + d];
  const float v1 = outs[((size_t)1 * B * L + b * L + (L - 1 - t)) * D_MODEL + d];
  x[i] += v0 + v1;
}

// --------------------------- small movers ----------------------------------
__global__ void copy_rows_kernel(const float* __restrict__ in,
                                 float* __restrict__ out,
                                 int B, int inL, int outL, int tOff) {
  const size_t total = (size_t)B * outL * D_MODEL;
  const size_t i = (size_t)blockIdx.x * blockDim.x + threadIdx.x;
  if (i >= total) return;
  const int d = (int)(i % D_MODEL);
  const size_t r = i / D_MODEL;
  const int t = (int)(r % outL);
  const size_t b = r / outL;
  out[i] = in[((size_t)b * inL + tOff + t) * D_MODEL + d];
}

__global__ void mean_pool_kernel(const float* __restrict__ in,
                                 float* __restrict__ out, int B, int L) {
  const int i = blockIdx.x * blockDim.x + threadIdx.x;
  if (i >= B * D_MODEL) return;
  const int d = i % D_MODEL;
  const int b = i / D_MODEL;
  float s = 0.f;
  for (int t = 0; t < L; ++t) s += in[((size_t)b * L + t) * D_MODEL + d];
  out[i] = s / (float)L;
}

__global__ void build_xfull_kernel(const float* __restrict__ xd,
                                   const float* __restrict__ mtok,
                                   const float* __restrict__ pos,
                                   float* __restrict__ xfull, int B) {
  const size_t total = (size_t)B * MAXT * D_MODEL;
  const size_t i = (size_t)blockIdx.x * blockDim.x + threadIdx.x;
  if (i >= total) return;
  const int d = (int)(i % D_MODEL);
  const size_t r = i / D_MODEL;
  const int t = (int)(r % MAXT);
  const size_t b = r / MAXT;
  const float base = (t < N_VIS) ? xd[((size_t)b * N_VIS + t) * D_MODEL + d]
                                 : mtok[d];
  xfull[i] = base + pos[(size_t)t * D_MODEL + d];
}

// --------------------------- host-side helpers -----------------------------
static inline void launch_gemm(hipStream_t s, const float* A, int lda,
                               const float* W, float* C,
                               int M, int N, int K, int zb,
                               long sA, long sW, long sC,
                               const float* bias, long sBias,
                               const float* aux, int auxT, int mode) {
  dim3 grid((N + 63) / 64, (M + 63) / 64, zb);
  gemm_wmma_kernel<<<grid, dim3(128), 0, s>>>(A, lda, W, C, M, N, K,
                                              sA, sW, sC, sBias,
                                              bias, aux, auxT, mode);
}

static void run_stack(hipStream_t s, float* x, int B, int L, int depth,
                      const float* lnw, const float* lnb, const float* ipw,
                      const float* cw,  const float* cb,  const float* xpw,
                      const float* dtw, const float* dtb, const float* Al,
                      const float* Dd,  const float* opw,
                      float* u, float* xz, float* xs, float* dbl,
                      float* dtpad, float* dt, float* ycomb, float* outs) {
  const int BL = B * L;
  const long szLn  = 2L * D_MODEL;
  const long szIp  = 2L * (2 * D_INNER) * D_MODEL;
  const long szCw  = 2L * D_INNER * D_CONV;
  const long szCb  = 2L * D_INNER;
  const long szXp  = 2L * (DT_RANK + 2 * D_STATE) * D_INNER;
  const long szDtw = 2L * D_INNER * DT_RANK;
  const long szDtb = 2L * D_INNER;
  const long szAl  = 2L * D_INNER * D_STATE;
  const long szDd  = 2L * D_INNER;
  const long szOp  = 2L * D_MODEL * D_INNER;

  for (int l = 0; l < depth; ++l) {
    const float* lnw_l = lnw + (size_t)l * szLn;
    const float* lnb_l = lnb + (size_t)l * szLn;
    const float* ipw_l = ipw + (size_t)l * szIp;
    const float* cw_l  = cw  + (size_t)l * szCw;
    const float* cb_l  = cb  + (size_t)l * szCb;
    const float* xpw_l = xpw + (size_t)l * szXp;
    const float* dtw_l = dtw + (size_t)l * szDtw;
    const float* dtb_l = dtb + (size_t)l * szDtb;
    const float* Al_l  = Al  + (size_t)l * szAl;
    const float* Dd_l  = Dd  + (size_t)l * szDd;
    const float* opw_l = opw + (size_t)l * szOp;

    // dual-direction LN (one wave / row)
    {
      int thr = 2 * BL * 32;
      ln_dual_kernel<<<(thr + 127) / 128, 128, 0, s>>>(x, u, lnw_l, lnb_l, B, L);
    }
    // in_proj: (2,BL,256) -> (2,BL,1024)
    launch_gemm(s, u, D_MODEL, ipw_l, xz, BL, 2 * D_INNER, D_MODEL, 2,
                (long)BL * D_MODEL, (long)(2 * D_INNER) * D_MODEL,
                (long)BL * 2 * D_INNER, nullptr, 0, nullptr, 0, 0);
    // causal conv + silu
    {
      size_t tc = (size_t)2 * BL * D_INNER;
      conv_silu_kernel<<<(unsigned)((tc + 255) / 256), 256, 0, s>>>(xz, xs, cw_l, cb_l, B, L);
    }
    // x_proj: (2,BL,512) -> (2,BL,48)
    launch_gemm(s, xs, D_INNER, xpw_l, dbl, BL, DT_RANK + 2 * D_STATE, D_INNER, 2,
                (long)BL * D_INNER, (long)(DT_RANK + 2 * D_STATE) * D_INNER,
                (long)BL * (DT_RANK + 2 * D_STATE), nullptr, 0, nullptr, 0, 0);
    // zero-pad dt_raw to K=32 then dt_proj + softplus
    {
      size_t tp = (size_t)2 * BL * 32;
      pad_dt_kernel<<<(unsigned)((tp + 255) / 256), 256, 0, s>>>(dbl, dtpad, 2 * BL);
    }
    launch_gemm(s, dtpad, 32, dtw_l, dt, BL, D_INNER, 32, 2,
                (long)BL * 32, (long)D_INNER * DT_RANK,
                (long)BL * D_INNER, dtb_l, (long)D_INNER, nullptr, 0, 1);
    // selective scan + gating
    {
      int st = 2 * B * D_INNER;
      scan_kernel<<<(st + 255) / 256, 256, 0, s>>>(dt, xs, dbl, xz, Al_l, Dd_l, ycomb, B, L);
    }
    // out_proj: (2,BL,512) -> (2,BL,256)
    launch_gemm(s, ycomb, D_INNER, opw_l, outs, BL, D_MODEL, D_INNER, 2,
                (long)BL * D_INNER, (long)D_MODEL * D_INNER,
                (long)BL * D_MODEL, nullptr, 0, nullptr, 0, 0);
    // x += out_fwd + flip(out_bwd)
    {
      size_t ce = (size_t)BL * D_MODEL;
      combine_kernel<<<(unsigned)((ce + 255) / 256), 256, 0, s>>>(x, outs, B, L);
    }
  }
}

// --------------------------- entry point -----------------------------------
extern "C" void kernel_launch(void* const* d_in, const int* in_sizes, int n_in,
                              void* d_out, int out_size, void* d_ws, size_t ws_size,
                              hipStream_t stream) {
  const float* x      = (const float*)d_in[0];
  // d_in[1] = mask (static: frames >= N_VIS) — indices are compile-time here.
  const float* pe_w   = (const float*)d_in[2];
  const float* pe_b   = (const float*)d_in[3];
  const float* pos    = (const float*)d_in[4];
  const float* e_lnw  = (const float*)d_in[5];
  const float* e_lnb  = (const float*)d_in[6];
  const float* e_ipw  = (const float*)d_in[7];
  const float* e_cw   = (const float*)d_in[8];
  const float* e_cb   = (const float*)d_in[9];
  const float* e_xpw  = (const float*)d_in[10];
  const float* e_dtw  = (const float*)d_in[11];
  const float* e_dtb  = (const float*)d_in[12];
  const float* e_Al   = (const float*)d_in[13];
  const float* e_Dd   = (const float*)d_in[14];
  const float* e_opw  = (const float*)d_in[15];
  const float* enc_nw = (const float*)d_in[16];
  const float* enc_nb = (const float*)d_in[17];
  const float* eh_w   = (const float*)d_in[18];
  const float* eh_b   = (const float*)d_in[19];
  const float* hash_w = (const float*)d_in[20];
  const float* hash_b = (const float*)d_in[21];
  const float* ed_w   = (const float*)d_in[22];
  const float* mtok   = (const float*)d_in[23];
  const float* d_lnw  = (const float*)d_in[24];
  const float* d_lnb  = (const float*)d_in[25];
  const float* d_ipw  = (const float*)d_in[26];
  const float* d_cw   = (const float*)d_in[27];
  const float* d_cb   = (const float*)d_in[28];
  const float* d_xpw  = (const float*)d_in[29];
  const float* d_dtw  = (const float*)d_in[30];
  const float* d_dtb  = (const float*)d_in[31];
  const float* d_Al   = (const float*)d_in[32];
  const float* d_Dd   = (const float*)d_in[33];
  const float* d_opw  = (const float*)d_in[34];
  const float* dn_w   = (const float*)d_in[35];
  const float* dn_b   = (const float*)d_in[36];
  const float* dh_w   = (const float*)d_in[37];
  const float* dh_b   = (const float*)d_in[38];

  float* recon = (float*)d_out;                                   // (64,13,4096)
  float* code  = recon + (size_t)BATCH * N_MASK * FEAT;           // (64,64)

  // ---- workspace carve-out (floats) ----
  float* W = (float*)d_ws;
  size_t off = 0;
  auto alloc = [&](size_t n) { float* p = W + off; off += n; return p; };
  float* h      = alloc((size_t)BATCH * MAXT * D_MODEL);          // 409600
  float* xcur   = alloc((size_t)BATCH * MAXT * D_MODEL);          // 409600
  float* u      = alloc((size_t)2 * BATCH * MAXT * D_MODEL);      // 819200
  float* xz     = alloc((size_t)2 * BATCH * MAXT * 2 * D_INNER);  // 3276800
  float* xs     = alloc((size_t)2 * BATCH * MAXT * D_INNER);      // 1638400
  float* dbl    = alloc((size_t)2 * BATCH * MAXT * (DT_RANK + 2 * D_STATE));
  float* dtpad  = alloc((size_t)2 * BATCH * MAXT * 32);           // 102400
  float* dt     = alloc((size_t)2 * BATCH * MAXT * D_INNER);      // 1638400
  float* ycomb  = alloc((size_t)2 * BATCH * MAXT * D_INNER);      // 1638400
  float* outs   = alloc((size_t)2 * BATCH * MAXT * D_MODEL);      // 819200
  float* lnout  = alloc((size_t)BATCH * N_MASK * D_MODEL);        // >= B*12*256
  float* xvise  = alloc((size_t)BATCH * N_VIS * D_MODEL);
  float* pooled = alloc((size_t)BATCH * D_MODEL);
  float* xd     = alloc((size_t)BATCH * N_VIS * D_MODEL);
  (void)ws_size; (void)in_sizes; (void)n_in; (void)out_size;

  // ---- 1. patch embed:  h = x @ pe_w^T + pe_b + pos ----
  launch_gemm(stream, x, FEAT, pe_w, h, BATCH * MAXT, D_MODEL, FEAT, 1,
              0, 0, 0, pe_b, 0, pos, MAXT, 3);

  // ---- 2. x_vis = h[:, :12] ----
  {
    size_t n = (size_t)BATCH * N_VIS * D_MODEL;
    copy_rows_kernel<<<(unsigned)((n + 255) / 256), 256, 0, stream>>>(
        h, xcur, BATCH, MAXT, N_VIS, 0);
  }

  // ---- 3. encoder (12 bidirectional mamba blocks, L=12) ----
  run_stack(stream, xcur, BATCH, N_VIS, ENC_D,
            e_lnw, e_lnb, e_ipw, e_cw, e_cb, e_xpw, e_dtw, e_dtb, e_Al, e_Dd,
            e_opw, u, xz, xs, dbl, dtpad, dt, ycomb, outs);

  // ---- 4. final encoder LN + eh projection ----
  {
    int waves = BATCH * N_VIS;
    ln_map_kernel<<<(waves * 32 + 127) / 128, 128, 0, stream>>>(
        xcur, lnout, enc_nw, enc_nb, BATCH, N_VIS, N_VIS, 0);
  }
  launch_gemm(stream, lnout, D_MODEL, eh_w, xvise, BATCH * N_VIS, D_MODEL,
              D_MODEL, 1, 0, 0, 0, eh_b, 0, nullptr, 0, 0);

  // ---- 5. hash code: rint(tanh(mean_t(xvise) @ hash_w^T + hash_b)) ----
  {
    int n = BATCH * D_MODEL;
    mean_pool_kernel<<<(n + 255) / 256, 256, 0, stream>>>(xvise, pooled, BATCH, N_VIS);
  }
  launch_gemm(stream, pooled, D_MODEL, hash_w, code, BATCH, NBITS, D_MODEL, 1,
              0, 0, 0, hash_b, 0, nullptr, 0, 2);

  // ---- 6. decoder input:  xd = xvise @ ed_w^T ; x_full = [xd+pos | mtok+pos] ----
  launch_gemm(stream, xvise, D_MODEL, ed_w, xd, BATCH * N_VIS, D_MODEL,
              D_MODEL, 1, 0, 0, 0, nullptr, 0, nullptr, 0, 0);
  {
    size_t n = (size_t)BATCH * MAXT * D_MODEL;
    build_xfull_kernel<<<(unsigned)((n + 255) / 256), 256, 0, stream>>>(
        xd, mtok, pos, xcur, BATCH);
  }

  // ---- 7. decoder (8 bidirectional mamba blocks, L=25) ----
  run_stack(stream, xcur, BATCH, MAXT, DEC_D,
            d_lnw, d_lnb, d_ipw, d_cw, d_cb, d_xpw, d_dtw, d_dtb, d_Al, d_Dd,
            d_opw, u, xz, xs, dbl, dtpad, dt, ycomb, outs);

  // ---- 8. tail LN + reconstruction head ----
  {
    int waves = BATCH * N_MASK;
    ln_map_kernel<<<(waves * 32 + 127) / 128, 128, 0, stream>>>(
        xcur, lnout, dn_w, dn_b, BATCH, MAXT, N_MASK, N_VIS);
  }
  launch_gemm(stream, lnout, D_MODEL, dh_w, recon, BATCH * N_MASK, FEAT,
              D_MODEL, 1, 0, 0, 0, dh_b, 0, nullptr, 0, 0);
}
